// ProbAttention_69698729279949
// MI455X (gfx1250) — matrix operand order repaired and checked by reading, compiled, not run on hardware
//
#include <hip/hip_runtime.h>
#include <hip/hip_bf16.h>

// ---------------------------------------------------------------------------
// ProbSparse attention (Informer) for MI455X / gfx1250, wave32 + WMMA.
// B=8, N=4096, C=512, H=8, Dh=64, u = sample_k = 45.
// GEMMs (91% of FLOPs): bf16 WMMA, 128x128 block tile, 64x32 wave tile,
// double-buffered LDS fed by GLOBAL_LOAD_ASYNC_TO_LDS_B128 (ASYNCcnt).
// ---------------------------------------------------------------------------

#define USE_ASYNC_LDS 1

#define BB   8
#define NN   4096
#define CC   512
#define HH   8
#define DH   64
#define USEL 45               // 5 * ceil(ln(4096)) = 45
#define MTOT (BB * NN)        // 32768 rows for the big GEMMs

typedef __attribute__((ext_vector_type(16))) __bf16 v16bf;
typedef __attribute__((ext_vector_type(8)))  float  v8f;

struct Bf16x16 { uint4 lo; uint4 hi; };   // 32 bytes == v16bf

__device__ __forceinline__ unsigned short f32_to_bf16(float f) {
    union { float f; unsigned u; } x; x.f = f;
    unsigned r = x.u + 0x7FFFu + ((x.u >> 16) & 1u);   // round-to-nearest-even
    return (unsigned short)(r >> 16);
}

// Async global->LDS copy of 16 bytes per lane (CDNA5 GLOBAL_LOAD_ASYNC_TO_LDS).
// Low 32 bits of a generic pointer to __shared__ are the LDS byte offset
// (ISA 10.2: LDS aperture address is truncated to addr[31:0]).
__device__ __forceinline__ void async_copy_b128(const void* g, void* l) {
#if USE_ASYNC_LDS
    unsigned laddr = (unsigned)(size_t)l;
    asm volatile("global_load_async_to_lds_b128 %0, %1, off"
                 :: "v"(laddr), "v"(g) : "memory");
#else
    *(uint4*)l = *(const uint4*)g;
#endif
}

__device__ __forceinline__ void async_wait_all() {
#if USE_ASYNC_LDS
    asm volatile("s_wait_asynccnt 0x0" ::: "memory");
#endif
}

// ---------------------------------------------------------------------------
// Generic f32 -> bf16 conversion
// ---------------------------------------------------------------------------
__global__ __launch_bounds__(256) void f32_to_bf16_kernel(
        const float* __restrict__ in, unsigned short* __restrict__ out, int n) {
    int i = blockIdx.x * 256 + threadIdx.x;
    if (i < n) out[i] = f32_to_bf16(in[i]);
}

// Wt[n][k] = bf16(W[k][n])  (weights pre-transposed so GEMM B-fragments are
// contiguous per lane in LDS)
__global__ __launch_bounds__(256) void transpose_w_bf16_kernel(
        const float* __restrict__ W, unsigned short* __restrict__ Wt) {
    int i = blockIdx.x * 256 + threadIdx.x;
    if (i < CC * CC) {
        int n = i / CC, k = i % CC;
        Wt[(size_t)n * CC + k] = f32_to_bf16(W[(size_t)k * CC + n]);
    }
}

// ---------------------------------------------------------------------------
// bf16 WMMA GEMM:  Out[M][Nc] = A[M][K](bf16) @ Bt[Nc][K](bf16)^T + bias
// Block tile 128x128, BK=32, 8 waves arranged 2(M) x 4(N), wave tile 64x32
// (4x2 WMMA 16x16x32 tiles -> 8 WMMAs per K-step per wave).
// Double-buffered LDS, async global->LDS staging.
// M % 128 == 0, Nc % 128 == 0, K % 64 == 0 at all call sites -> no guards,
// EXEC is all-ones at every WMMA (ISA requirement).
// ---------------------------------------------------------------------------
__global__ __launch_bounds__(256) void gemm_bf16_wmma_kernel(
        const unsigned short* __restrict__ A,
        const unsigned short* __restrict__ Bt,
        const float* __restrict__ bias,
        float* __restrict__ Out,
        int Mrows, int Ncols, int K) {
    __shared__ unsigned short lA[2][128 * 32];   // 2 x 8 KB ping-pong
    __shared__ unsigned short lB[2][128 * 32];   // 2 x 8 KB ping-pong

    const int tid  = threadIdx.x;
    const int wave = tid >> 5;
    const int lane = tid & 31;
    const int wmi  = wave & 1;                // 0..1  (M direction, 64 rows)
    const int wni  = wave >> 1;               // 0..3  (N direction, 32 cols)
    const int row0 = blockIdx.y * 128;
    const int col0 = blockIdx.x * 128;
    const int mrow = lane & 15;
    const int half = lane >> 4;

    v8f acc[4][2] = {};

    // staging: 128 rows x 32 halves per tile; thread t covers row t>>1,
    // halves [16*(t&1), 16*(t&1)+15]  (two b128 per thread per matrix)
    const int sr = tid >> 1, ss = tid & 1;

    auto stage = [&](int buf, int kk) {
        const unsigned short* ga = &A[(size_t)(row0 + sr) * K + kk + 16 * ss];
        unsigned short*       la = &lA[buf][sr * 32 + 16 * ss];
        async_copy_b128(ga,     la);
        async_copy_b128(ga + 8, la + 8);
        const unsigned short* gb = &Bt[(size_t)(col0 + sr) * K + kk + 16 * ss];
        unsigned short*       lb = &lB[buf][sr * 32 + 16 * ss];
        async_copy_b128(gb,     lb);
        async_copy_b128(gb + 8, lb + 8);
    };

    // fragments per ISA 7.12.2 16-bit layouts
    auto compute = [&](int buf) {
        v16bf bfr[2];
#pragma unroll
        for (int ni = 0; ni < 2; ++ni) {
            const unsigned short* p =
                &lB[buf][(wni * 32 + ni * 16 + mrow) * 32 + 8 * half];
            Bf16x16 t;
            t.lo = *(const uint4*)p;          // k = 8*half .. +7
            t.hi = *(const uint4*)(p + 16);   // k = 16+8*half .. +7
            bfr[ni] = __builtin_bit_cast(v16bf, t);
        }
#pragma unroll
        for (int mi = 0; mi < 4; ++mi) {
            const unsigned short* pa =
                &lA[buf][(wmi * 64 + mi * 16 + mrow) * 32 + 8 * half];
            Bf16x16 t;
            t.lo = *(const uint4*)pa;         // k = 8*half .. +7
            t.hi = *(const uint4*)(pa + 16);  // k = 16+8*half .. +7
            v16bf af = __builtin_bit_cast(v16bf, t);
#pragma unroll
            for (int ni = 0; ni < 2; ++ni)
                acc[mi][ni] = __builtin_amdgcn_wmma_f32_16x16x32_bf16(
                        false, af, false, bfr[ni],
                        (short)0, acc[mi][ni], false, false);
        }
    };

    // prologue: fill buffer 0
    stage(0, 0);
    async_wait_all();
    __syncthreads();

    for (int k0 = 0; k0 < K; k0 += 64) {
        bool n1 = (k0 + 32) < K;
        if (n1) stage(1, k0 + 32);            // overlap with compute(0)
        compute(0);
        if (n1) async_wait_all();
        __syncthreads();
        if (n1) {
            bool n2 = (k0 + 64) < K;
            if (n2) stage(0, k0 + 64);        // overlap with compute(1)
            compute(1);
            if (n2) async_wait_all();
            __syncthreads();
        }
    }

    // epilogue: C/D layout VGPR r -> M = r + 8*half, lane&15 -> N
#pragma unroll
    for (int mi = 0; mi < 4; ++mi) {
#pragma unroll
        for (int ni = 0; ni < 2; ++ni) {
            int n = col0 + wni * 32 + ni * 16 + (lane & 15);
            float bsv = bias ? bias[n] : 0.0f;
#pragma unroll
            for (int r = 0; r < 8; ++r) {
                int m = row0 + wmi * 64 + mi * 16 + r + 8 * half;
                Out[(size_t)m * Ncols + n] = acc[mi][ni][r] + bsv;
            }
        }
    }
}

// ---------------------------------------------------------------------------
// M[b,h,l] = max_s(q.k_sample) - sum_s(q.k_sample)/N  (one wave per row)
// q,k stored as [B,N,C] f32 (head h is columns h*64..h*64+63).
// ---------------------------------------------------------------------------
__global__ __launch_bounds__(256) void sample_score_kernel(
        const float* __restrict__ q, const float* __restrict__ k,
        const int* __restrict__ idx, float* __restrict__ Ms) {
    int wid  = blockIdx.x * 8 + (threadIdx.x >> 5);
    int lane = threadIdx.x & 31;
    if (wid >= BB * HH * NN) return;
    int l  = wid % NN;
    int bh = wid / NN;
    int h  = bh % HH, b = bh / HH;
    const float* qrow = q + ((size_t)(b * NN + l) * CC + h * DH);
    float mx = -1e30f, sm = 0.0f;
    for (int s = lane; s < USEL; s += 32) {
        int kl = idx[l * USEL + s];
        const float* krow = k + ((size_t)(b * NN + kl) * CC + h * DH);
        float d = 0.0f;
#pragma unroll 8
        for (int t = 0; t < DH; ++t) d += qrow[t] * krow[t];
        mx = fmaxf(mx, d);
        sm += d;
    }
    for (int off = 16; off; off >>= 1) {
        mx = fmaxf(mx, __shfl_xor(mx, off));
        sm += __shfl_xor(sm, off);
    }
    if (lane == 0) Ms[(size_t)bh * NN + l] = mx - sm / (float)NN;
}

// ---------------------------------------------------------------------------
// top-45 indices per (b,h): iterative argmax over 4096 scores in LDS
// ---------------------------------------------------------------------------
__global__ __launch_bounds__(256) void topk_kernel(
        const float* __restrict__ Ms, int* __restrict__ Mtop) {
    __shared__ float vals[NN];       // 16 KB
    __shared__ float rmax[256];
    __shared__ int   ridx[256];
    int bh = blockIdx.x, tid = threadIdx.x;
    const float* src = Ms + (size_t)bh * NN;
    for (int i = tid; i < NN; i += 256) vals[i] = src[i];
    __syncthreads();
    for (int it = 0; it < USEL; ++it) {
        float bm = -1e30f; int bi = 0;
        for (int i = tid; i < NN; i += 256)
            if (vals[i] > bm) { bm = vals[i]; bi = i; }
        rmax[tid] = bm; ridx[tid] = bi;
        __syncthreads();
        for (int s = 128; s; s >>= 1) {
            if (tid < s && rmax[tid + s] > rmax[tid]) {
                rmax[tid] = rmax[tid + s]; ridx[tid] = ridx[tid + s];
            }
            __syncthreads();
        }
        if (tid == 0) { Mtop[bh * USEL + it] = ridx[0]; vals[ridx[0]] = -1e30f; }
        __syncthreads();
    }
}

// ---------------------------------------------------------------------------
// v_mean[b,h,d] = mean_l v[b,l,h*64+d]
// ---------------------------------------------------------------------------
__global__ __launch_bounds__(256) void vmean_kernel(
        const float* __restrict__ v, float* __restrict__ vmean) {
    __shared__ float red[256];
    int bh = blockIdx.x;
    int h = bh % HH, b = bh / HH;
    int tid = threadIdx.x, d = tid & 63, c = tid >> 6;
    float s = 0.0f;
    for (int l = c * (NN / 4); l < (c + 1) * (NN / 4); ++l)
        s += v[(size_t)(b * NN + l) * CC + h * DH + d];
    red[tid] = s;
    __syncthreads();
    if (tid < DH)
        vmean[bh * DH + tid] =
            (red[tid] + red[64 + tid] + red[128 + tid] + red[192 + tid]) *
            (1.0f / NN);
}

// ctx[b,n,c] = v_mean[b, c/64, c%64]   (broadcast baseline)
__global__ __launch_bounds__(256) void ctx_init_kernel(
        const float* __restrict__ vmean, float* __restrict__ ctx) {
    size_t i = (size_t)blockIdx.x * 256 + threadIdx.x;
    if (i < (size_t)BB * NN * CC) {
        int c = (int)(i % CC);
        int b = (int)(i / ((size_t)NN * CC));
        ctx[i] = vmean[(b * HH + (c >> 6)) * DH + (c & 63)];
    }
}

// ---------------------------------------------------------------------------
// For each of the 45 selected rows per (b,h): full softmax attention over all
// 4096 keys; overwrite context row.  One block per (b,h,u).
// ---------------------------------------------------------------------------
__global__ __launch_bounds__(256) void prob_attn_kernel(
        const float* __restrict__ q, const float* __restrict__ k,
        const float* __restrict__ v, const int* __restrict__ Mtop,
        float* __restrict__ ctx) {
    __shared__ float sc[NN];         // 16 KB of scores/probs
    __shared__ float qrow[DH];
    __shared__ float red[256];
    int u  = blockIdx.x % USEL;
    int bh = blockIdx.x / USEL;
    int h = bh % HH, b = bh / HH;
    int l = Mtop[bh * USEL + u];
    int tid = threadIdx.x;
    if (tid < DH) qrow[tid] = q[(size_t)(b * NN + l) * CC + h * DH + tid];
    __syncthreads();

    const float scale = 0.125f;      // 1/sqrt(64)
    float mx = -1e30f;
    for (int kk = tid; kk < NN; kk += 256) {
        const float* krow = k + ((size_t)(b * NN + kk) * CC + h * DH);
        float d = 0.0f;
#pragma unroll 8
        for (int t = 0; t < DH; ++t) d += qrow[t] * krow[t];
        d *= scale;
        sc[kk] = d;
        mx = fmaxf(mx, d);
    }
    red[tid] = mx; __syncthreads();
    for (int s = 128; s; s >>= 1) {
        if (tid < s) red[tid] = fmaxf(red[tid], red[tid + s]);
        __syncthreads();
    }
    mx = red[0];
    __syncthreads();

    float sum = 0.0f;
    for (int kk = tid; kk < NN; kk += 256) {
        float e = __expf(sc[kk] - mx);
        sc[kk] = e;
        sum += e;
    }
    red[tid] = sum; __syncthreads();
    for (int s = 128; s; s >>= 1) {
        if (tid < s) red[tid] += red[tid + s];
        __syncthreads();
    }
    float inv = 1.0f / red[0];
    __syncthreads();

    int d = tid & 63, c = tid >> 6;
    float acc = 0.0f;
    for (int kk = c * (NN / 4); kk < (c + 1) * (NN / 4); ++kk)
        acc += sc[kk] * v[(size_t)(b * NN + kk) * CC + h * DH + d];
    red[tid] = acc; __syncthreads();
    if (tid < DH)
        ctx[(size_t)(b * NN + l) * CC + h * DH + tid] =
            (red[tid] + red[64 + tid] + red[128 + tid] + red[192 + tid]) * inv;
}

// ---------------------------------------------------------------------------
extern "C" void kernel_launch(void* const* d_in, const int* in_sizes, int n_in,
                              void* d_out, int out_size, void* d_ws, size_t ws_size,
                              hipStream_t stream) {
    const float* x  = (const float*)d_in[0];
    const float* Wq = (const float*)d_in[1];
    const float* bq = (const float*)d_in[2];
    const float* Wk = (const float*)d_in[3];
    const float* bk = (const float*)d_in[4];
    const float* Wv = (const float*)d_in[5];
    const float* bv = (const float*)d_in[6];
    const float* Wp = (const float*)d_in[7];
    const float* bp = (const float*)d_in[8];
    const int*  idx = (const int*)d_in[9];   // int32 (JAX default, no x64)

    // ---- carve workspace ----
    char* p = (char*)d_ws;
    auto carve = [&](size_t bytes) -> char* {
        char* r = p;
        p += (bytes + 255) & ~(size_t)255;
        return r;
    };
    unsigned short* xb   = (unsigned short*)carve((size_t)MTOT * CC * 2);
    unsigned short* Wqb  = (unsigned short*)carve((size_t)CC * CC * 2);
    unsigned short* Wkb  = (unsigned short*)carve((size_t)CC * CC * 2);
    unsigned short* Wvb  = (unsigned short*)carve((size_t)CC * CC * 2);
    unsigned short* Wpb  = (unsigned short*)carve((size_t)CC * CC * 2);
    float*          qf   = (float*)carve((size_t)MTOT * CC * 4);
    float*          kf   = (float*)carve((size_t)MTOT * CC * 4);
    float*          vf   = (float*)carve((size_t)MTOT * CC * 4);
    float*          ctx  = (float*)carve((size_t)MTOT * CC * 4);
    unsigned short* ctxb = (unsigned short*)carve((size_t)MTOT * CC * 2);
    float*          Ms   = (float*)carve((size_t)BB * HH * NN * 4);
    int*            Mtop = (int*)carve((size_t)BB * HH * USEL * 4);
    float*          vmn  = (float*)carve((size_t)BB * HH * DH * 4);

    const int nElem   = MTOT * CC;                 // 16,777,216
    const int nBlocks = nElem / 256;               // 65,536
    dim3 gemmGrid(CC / 128, MTOT / 128);           // (4, 256)

    // 1) convert inputs to bf16 (x row-major, weights transposed)
    f32_to_bf16_kernel<<<nBlocks, 256, 0, stream>>>(x, xb, nElem);
    transpose_w_bf16_kernel<<<(CC * CC) / 256, 256, 0, stream>>>(Wq, Wqb);
    transpose_w_bf16_kernel<<<(CC * CC) / 256, 256, 0, stream>>>(Wk, Wkb);
    transpose_w_bf16_kernel<<<(CC * CC) / 256, 256, 0, stream>>>(Wv, Wvb);
    transpose_w_bf16_kernel<<<(CC * CC) / 256, 256, 0, stream>>>(Wp, Wpb);

    // 2) q,k,v projections (bf16 WMMA, f32 out)
    gemm_bf16_wmma_kernel<<<gemmGrid, 256, 0, stream>>>(xb, Wqb, bq, qf, MTOT, CC, CC);
    gemm_bf16_wmma_kernel<<<gemmGrid, 256, 0, stream>>>(xb, Wkb, bk, kf, MTOT, CC, CC);
    gemm_bf16_wmma_kernel<<<gemmGrid, 256, 0, stream>>>(xb, Wvb, bv, vf, MTOT, CC, CC);

    // 3) sampled scores + top-45 per (b,h)
    sample_score_kernel<<<(BB * HH * NN) / 8, 256, 0, stream>>>(qf, kf, idx, Ms);
    topk_kernel<<<BB * HH, 256, 0, stream>>>(Ms, Mtop);

    // 4) baseline context = broadcast v-mean; overwrite selected rows
    vmean_kernel<<<BB * HH, 256, 0, stream>>>(vf, vmn);
    ctx_init_kernel<<<nBlocks, 256, 0, stream>>>(vmn, ctx);
    prob_attn_kernel<<<BB * HH * USEL, 256, 0, stream>>>(qf, kf, vf, Mtop, ctx);

    // 5) output projection (bf16 WMMA) -> d_out (f32)
    f32_to_bf16_kernel<<<nBlocks, 256, 0, stream>>>(ctx, ctxb, nElem);
    gemm_bf16_wmma_kernel<<<gemmGrid, 256, 0, stream>>>(ctxb, Wpb, bp,
                                                        (float*)d_out, MTOT, CC, CC);
}
